// GraphBased_selfAttnLayer_20667382628453
// MI455X (gfx1250) — matrix operand order repaired and checked by reading, not compiled, not run on hardware
//
#include <hip/hip_runtime.h>

// ============================================================================
// Fused self-attention-ish layer for MI455X (gfx1250, wave32, WMMA).
//
// Pipeline:
//   1) bf16 conversions of x, wq, wk, wv, m1w, m2w
//   2) q,k,v = x@W^T + b           (WMMA bf16 -> f32)
//   3) Frobenius norms of q,k      (block reduce + global f32 atomics)
//   4) q,k -> bf16; v -> v^T bf16  (tiled transpose)
//   5) FUSED: attn = conv3x3((q@k^T)/(|q||k|)) @ v   -- A never hits HBM.
//      64-row bands, 32-col stripes, 3-stripe LDS ring for the conv halo,
//      GEMM2 accumulated in registers over 256 stripes.
//   6) h = x + LN(attn); lh = LN(h) (bf16)
//   7) z = lh @ m1w^T (WMMA)
//   8) batchnorm stats over rows (per-column scale/shift)
//   9) out = h + relu(bn(z)) @ m2w^T (WMMA, bn+relu fused into A-frag build)
// ============================================================================

#define N_TOK 8192
#define DMODEL 256
#define HDIM 128
#define EPS 1e-5f

typedef __bf16 bf16;
typedef __attribute__((ext_vector_type(16))) __bf16 bf16x16;
typedef __attribute__((ext_vector_type(8)))  __bf16 bf16x8;
typedef __attribute__((ext_vector_type(8)))  float  f32x8;

#define WMMA_BF16(A, B, C) \
  __builtin_amdgcn_wmma_f32_16x16x32_bf16(false, (A), false, (B), (short)0, (C), false, false)

// ---- WMMA fragment helpers (wave32 layouts per CDNA5 ISA 7.12.2) -----------

// A-matrix 16x32 bf16, row-major source. lanes 0-15: M=lane, K base 0;
// lanes 16-31: M=lane-16, K base 8. Chunks [kb..kb+7] and [kb+16..kb+23].
__device__ __forceinline__ bf16x16 load_afrag(const bf16* base, int ld, int m0, int k0) {
  int lane = threadIdx.x & 31;
  const bf16* p = base + (size_t)(m0 + (lane & 15)) * ld + k0 + ((lane & 16) ? 8 : 0);
  union { bf16x16 v; bf16x8 h[2]; } u;
  u.h[0] = *(const bf16x8*)p;
  u.h[1] = *(const bf16x8*)(p + 16);
  return u.v;
}

// B-matrix 32x16 where B[k][n] = S[n][k] with S row-major (i.e. C = A @ S^T).
// lanes 0-15: N=lane, K base 0; lanes 16-31: N=lane-16, K base 8. Chunks [kb..kb+7] and [kb+16..kb+23].
__device__ __forceinline__ bf16x16 load_bfragT(const bf16* S, int ld, int n0, int k0) {
  int lane = threadIdx.x & 31;
  const bf16* p = S + (size_t)(n0 + (lane & 15)) * ld + k0 + ((lane & 16) ? 8 : 0);
  union { bf16x16 v; bf16x8 h[2]; } u;
  u.h[0] = *(const bf16x8*)p;
  u.h[1] = *(const bf16x8*)(p + 16);
  return u.v;
}

// C/D 16x16 f32: VGPR r -> row M = r + (lane<16 ? 0 : 8), col N = lane&15.
__device__ __forceinline__ void store_cfrag(float* C, int ld, int m0, int n0, f32x8 acc) {
  int lane = threadIdx.x & 31;
  int n = n0 + (lane & 15);
  int mb = m0 + ((lane & 16) ? 8 : 0);
#pragma unroll
  for (int r = 0; r < 8; ++r) C[(size_t)(mb + r) * ld + n] = acc[r];
}

// ---- small utility kernels -------------------------------------------------

__global__ void k_f32_to_bf16(const float* __restrict__ in, bf16* __restrict__ out, int n) {
  int i = blockIdx.x * blockDim.x + threadIdx.x;
  if (i < n) out[i] = (bf16)in[i];
}

__global__ void k_zero_f32(float* p, int n) {
  int i = blockIdx.x * blockDim.x + threadIdx.x;
  if (i < n) p[i] = 0.f;
}

__global__ __launch_bounds__(256) void k_sumsq(const float* __restrict__ p, int n, float* acc) {
  __shared__ float red[256];
  float s = 0.f;
  for (int i = blockIdx.x * 256 + threadIdx.x; i < n; i += gridDim.x * 256) s += p[i] * p[i];
  red[threadIdx.x] = s;
  __syncthreads();
  for (int off = 128; off > 0; off >>= 1) {
    if (threadIdx.x < off) red[threadIdx.x] += red[threadIdx.x + off];
    __syncthreads();
  }
  if (threadIdx.x == 0) atomicAdd(acc, red[0]);
}

// ---- projections: out(N x 256) = xb(N x 256) @ wb^T + bias -----------------
__global__ __launch_bounds__(256) void k_proj(const bf16* __restrict__ xb,
                                              const bf16* __restrict__ wb,
                                              const float* __restrict__ bias,
                                              float* __restrict__ out) {
  int wave = threadIdx.x >> 5, lane = threadIdx.x & 31;
  int m0 = blockIdx.x * 16;
  int n0 = wave * 32;  // 8 waves x 32 cols = 256
  f32x8 c0 = {}, c1 = {};
#pragma unroll
  for (int kc = 0; kc < 8; ++kc) {
    bf16x16 a = load_afrag(xb, DMODEL, m0, kc * 32);
    bf16x16 b0 = load_bfragT(wb, DMODEL, n0, kc * 32);
    bf16x16 b1 = load_bfragT(wb, DMODEL, n0 + 16, kc * 32);
    c0 = WMMA_BF16(a, b0, c0);
    c1 = WMMA_BF16(a, b1, c1);
  }
  int n = lane & 15;
  int mb = m0 + ((lane & 16) ? 8 : 0);
  float bb0 = bias[n0 + n], bb1 = bias[n0 + 16 + n];
#pragma unroll
  for (int r = 0; r < 8; ++r) {
    out[(size_t)(mb + r) * DMODEL + n0 + n] = c0[r] + bb0;
    out[(size_t)(mb + r) * DMODEL + n0 + 16 + n] = c1[r] + bb1;
  }
}

// ---- v (N x 256, f32) -> vT (256 x N, bf16) tiled transpose ----------------
__global__ __launch_bounds__(256) void k_transpose_bf16(const float* __restrict__ v,
                                                        bf16* __restrict__ vT) {
  __shared__ float tile[32][33];
  int tx = threadIdx.x & 31, ty = threadIdx.x >> 5;
  int nb = blockIdx.x, db = blockIdx.y;
#pragma unroll
  for (int i = 0; i < 32; i += 8)
    tile[ty + i][tx] = v[(size_t)(nb * 32 + ty + i) * DMODEL + db * 32 + tx];
  __syncthreads();
#pragma unroll
  for (int i = 0; i < 32; i += 8)
    vT[(size_t)(db * 32 + ty + i) * N_TOK + nb * 32 + tx] = (bf16)tile[tx][ty + i];
}

// ---- fused A = conv3x3((q k^T)*inv) ; attn = A @ v -------------------------
#define BAND 80     // computed band rows (5 M-tiles), global rows m0-1 .. m0+78
#define QS_LD 264   // q staging stride (pad: 264*2B = 528B = 16B aligned, bank-safe)
#define SA_ROWS 66  // band rows actually consumed by conv (m0-1 .. m0+64)
#define SA_LD 33    // f32 stripe stride (odd -> conflict-free column reads)
#define SC_LD 40    // conv-out bf16 stride (80B = 16B aligned, bank-safe b128)
#define NSTRIPE 256 // 8192 / 32

__device__ __forceinline__ void conv_gemm2(int sc, int tid, int tm, int nb0,
                                           const float (*sA)[SA_ROWS * SA_LD],
                                           bf16* sConv, const float* scw, float cbias,
                                           const bf16* vT, f32x8* acc) {
  int cur = sc % 3, prv = (sc + 2) % 3, nxt = (sc + 1) % 3;
  // conv: thread -> output row r (0..63 == band row r+1), 8 columns
  int r = tid & 63, c0 = (tid >> 6) * 8;
#pragma unroll
  for (int i = 0; i < 8; ++i) {
    int c = c0 + i;
    float s = cbias;  // conv bias folded into A before the contraction
#pragma unroll
    for (int dr = 0; dr < 3; ++dr) {
      int ro = (r + dr) * SA_LD;  // band rows r..r+2
#pragma unroll
      for (int dc = 0; dc < 3; ++dc) {
        int cc = c - 1 + dc;
        float a;
        if (cc < 0)       a = sA[prv][ro + cc + 32];
        else if (cc > 31) a = sA[nxt][ro + cc - 32];
        else              a = sA[cur][ro + cc];
        s += scw[dr * 3 + dc] * a;
      }
    }
    sConv[r * SC_LD + c] = (bf16)s;
  }
  __syncthreads();
  // GEMM2: acc += convA(64x32) @ v(32 x 256), this wave: Mtile tm, Ntiles nb0..nb0+7
  bf16x16 a = load_afrag(sConv, SC_LD, 16 * tm, 0);
#pragma unroll
  for (int nt = 0; nt < 8; ++nt) {
    bf16x16 b = load_bfragT(vT, N_TOK, (nb0 + nt) * 16, 32 * sc);
    acc[nt] = WMMA_BF16(a, b, acc[nt]);
  }
}

__global__ __launch_bounds__(256) void k_fused_attn(const bf16* __restrict__ qb,
                                                    const bf16* __restrict__ kb,
                                                    const bf16* __restrict__ vT,
                                                    const float* __restrict__ sqp,
                                                    const float* __restrict__ skp,
                                                    const float* __restrict__ cw,
                                                    const float* __restrict__ cb,
                                                    float* __restrict__ attn) {
  __shared__ __align__(16) bf16 qs[BAND * QS_LD];        // ~42.2 KB
  __shared__ float sA[3][SA_ROWS * SA_LD];               // ~26.1 KB (f32 ring)
  __shared__ __align__(16) bf16 sConv[64 * SC_LD];       // 5 KB
  __shared__ float scw[9];

  int tid = threadIdx.x;
  int wave = tid >> 5, lane = tid & 31;
  int m0 = blockIdx.x * 64;

  if (tid < 9) scw[tid] = cw[tid];
  float cbias = cb[0];
  float inv = rsqrtf(sqp[0] * skp[0]);  // 1/(|q|_F * |k|_F)

  // stage q band once; OOB rows zero-filled == conv's SAME zero padding
  for (int idx = tid; idx < BAND * 32; idx += 256) {
    int row = idx >> 5, ch = idx & 31;
    int gm = m0 - 1 + row;
    bf16x8 val = {};
    if (gm >= 0 && gm < N_TOK) val = *(const bf16x8*)(qb + (size_t)gm * DMODEL + ch * 8);
    *(bf16x8*)(qs + row * QS_LD + ch * 8) = val;
  }
  for (int idx = tid; idx < SA_ROWS * SA_LD; idx += 256) sA[2][idx] = 0.f;  // stripe -1
  __syncthreads();

  f32x8 acc[8] = {};
  int tm = wave & 3, nb0 = (wave >> 2) * 8;  // wave -> Mtile + 8 Ntiles of attn

  for (int s = 0; s < NSTRIPE; ++s) {
    int ring = s % 3;
    // GEMM1: A stripe s = q_band @ k[32s..32s+31]^T, 5 Mtiles x 2 coltiles
    for (int j = wave; j < 10; j += 8) {
      int t = j % 5, ct = j / 5;
      f32x8 c = {};
#pragma unroll
      for (int kc = 0; kc < 8; ++kc) {
        bf16x16 a = load_afrag(qs, QS_LD, 16 * t, kc * 32);               // LDS
        bf16x16 b = load_bfragT(kb, DMODEL, 32 * s + 16 * ct, kc * 32);   // L2-resident
        c = WMMA_BF16(a, b, c);
      }
      int n = lane & 15, rb = (lane & 16) ? 8 : 0;
#pragma unroll
      for (int r = 0; r < 8; ++r) {
        int br = 16 * t + rb + r;
        if (br < SA_ROWS) sA[ring][br * SA_LD + 16 * ct + n] = c[r] * inv;
      }
    }
    if (s + 1 < NSTRIPE) __builtin_prefetch(kb + (size_t)(32 * (s + 1)) * DMODEL, 0, 1);
    __syncthreads();
    if (s >= 1) conv_gemm2(s - 1, tid, tm, nb0, sA, sConv, scw, cbias, vT, acc);
  }
  // tail: last stripe sees zeros as its "next" stripe (right edge padding)
  for (int idx = tid; idx < SA_ROWS * SA_LD; idx += 256) sA[NSTRIPE % 3][idx] = 0.f;
  __syncthreads();
  conv_gemm2(NSTRIPE - 1, tid, tm, nb0, sA, sConv, scw, cbias, vT, acc);

#pragma unroll
  for (int nt = 0; nt < 8; ++nt)
    store_cfrag(attn, DMODEL, m0 + 16 * tm, (nb0 + nt) * 16, acc[nt]);
}

// ---- h = x + LN(attn); lh = LN(h) (bf16) -----------------------------------
__device__ __forceinline__ float block_sum(float v, float* red) {
  int t = threadIdx.x;
  red[t] = v;
  __syncthreads();
  for (int off = 128; off > 0; off >>= 1) {
    if (t < off) red[t] += red[t + off];
    __syncthreads();
  }
  float s = red[0];
  __syncthreads();
  return s;
}

__global__ __launch_bounds__(256) void k_ln(const float* __restrict__ x,
                                            const float* __restrict__ attn,
                                            const float* __restrict__ g1, const float* __restrict__ b1,
                                            const float* __restrict__ g2, const float* __restrict__ b2,
                                            float* __restrict__ h, bf16* __restrict__ lhb) {
  __shared__ float red[256];
  int row = blockIdx.x, t = threadIdx.x;
  float a = attn[(size_t)row * DMODEL + t];
  float m = block_sum(a, red) * (1.f / DMODEL);
  float d = a - m;
  float var = block_sum(d * d, red) * (1.f / DMODEL);
  float hv = x[(size_t)row * DMODEL + t] + d * rsqrtf(var + EPS) * g1[t] + b1[t];
  h[(size_t)row * DMODEL + t] = hv;
  float m2 = block_sum(hv, red) * (1.f / DMODEL);
  float d2 = hv - m2;
  float v2 = block_sum(d2 * d2, red) * (1.f / DMODEL);
  lhb[(size_t)row * DMODEL + t] = (bf16)(d2 * rsqrtf(v2 + EPS) * g2[t] + b2[t]);
}

// ---- z(N x 128) = lh @ m1w^T ----------------------------------------------
__global__ __launch_bounds__(256) void k_mlp1(const bf16* __restrict__ lhb,
                                              const bf16* __restrict__ m1wb,
                                              float* __restrict__ z) {
  int wave = threadIdx.x >> 5;
  int m0 = blockIdx.x * 16, n0 = wave * 16;  // 8 waves x 16 = 128 cols
  f32x8 c = {};
#pragma unroll
  for (int kc = 0; kc < 8; ++kc) {
    bf16x16 a = load_afrag(lhb, DMODEL, m0, kc * 32);
    bf16x16 b = load_bfragT(m1wb, DMODEL, n0, kc * 32);
    c = WMMA_BF16(a, b, c);
  }
  store_cfrag(z, HDIM, m0, n0, c);
}

// ---- batchnorm stats over rows: per-column scale/shift ---------------------
__global__ __launch_bounds__(256) void k_bnstats(const float* __restrict__ z,
                                                 const float* __restrict__ bng,
                                                 const float* __restrict__ bnb,
                                                 float* __restrict__ scale,
                                                 float* __restrict__ shift) {
  __shared__ float red[256];
  int c = blockIdx.x, t = threadIdx.x;
  float s = 0.f, s2 = 0.f;
  for (int r = t; r < N_TOK; r += 256) {
    float v = z[(size_t)r * HDIM + c];
    s += v;
    s2 += v * v;
  }
  float S = block_sum(s, red);
  float S2 = block_sum(s2, red);
  if (t == 0) {
    float mu = S * (1.f / N_TOK);
    float var = S2 * (1.f / N_TOK) - mu * mu;
    float sc = bng[c] * rsqrtf(var + EPS);
    scale[c] = sc;
    shift[c] = bnb[c] - mu * sc;
  }
}

// ---- out = h + relu(bn(z)) @ m2w^T (bn+relu fused into A-frags) ------------
__global__ __launch_bounds__(256) void k_mlp2(const float* __restrict__ z,
                                              const float* __restrict__ h,
                                              const bf16* __restrict__ m2wb,
                                              const float* __restrict__ scale,
                                              const float* __restrict__ shift,
                                              float* __restrict__ out) {
  __shared__ float ssc[HDIM], ssh[HDIM];
  int tid = threadIdx.x, wave = tid >> 5, lane = tid & 31;
  if (tid < HDIM) { ssc[tid] = scale[tid]; ssh[tid] = shift[tid]; }
  __syncthreads();
  int m0 = blockIdx.x * 16;
  int n0 = wave * 32;
  f32x8 c0 = {}, c1 = {};
  int mrow = m0 + (lane & 15);
#pragma unroll
  for (int kc = 0; kc < 4; ++kc) {  // K = 128
    int kb0 = kc * 32 + ((lane & 16) ? 8 : 0);
    union { bf16x16 v; bf16 e[16]; } u;
#pragma unroll
    for (int i = 0; i < 8; ++i) {
      int kk = kb0 + i;
      u.e[i] = (bf16)fmaxf(z[(size_t)mrow * HDIM + kk] * ssc[kk] + ssh[kk], 0.f);
      int kk2 = kb0 + 16 + i;
      u.e[8 + i] = (bf16)fmaxf(z[(size_t)mrow * HDIM + kk2] * ssc[kk2] + ssh[kk2], 0.f);
    }
    bf16x16 b0 = load_bfragT(m2wb, HDIM, n0, kc * 32);
    bf16x16 b1 = load_bfragT(m2wb, HDIM, n0 + 16, kc * 32);
    c0 = WMMA_BF16(u.v, b0, c0);
    c1 = WMMA_BF16(u.v, b1, c1);
  }
  int n = lane & 15, mb = m0 + ((lane & 16) ? 8 : 0);
#pragma unroll
  for (int r = 0; r < 8; ++r) {
    size_t i0 = (size_t)(mb + r) * DMODEL + n0 + n;
    size_t i1 = (size_t)(mb + r) * DMODEL + n0 + 16 + n;
    out[i0] = h[i0] + c0[r];
    out[i1] = h[i1] + c1[r];
  }
}

// ============================================================================
extern "C" void kernel_launch(void* const* d_in, const int* in_sizes, int n_in,
                              void* d_out, int out_size, void* d_ws, size_t ws_size,
                              hipStream_t stream) {
  (void)in_sizes; (void)n_in; (void)out_size; (void)ws_size;
  const float* x    = (const float*)d_in[0];
  const float* wq   = (const float*)d_in[1];
  const float* bq   = (const float*)d_in[2];
  const float* wk   = (const float*)d_in[3];
  const float* bk   = (const float*)d_in[4];
  const float* wv   = (const float*)d_in[5];
  const float* bv   = (const float*)d_in[6];
  const float* cw   = (const float*)d_in[7];
  const float* cb   = (const float*)d_in[8];
  const float* m1w  = (const float*)d_in[9];
  const float* m2w  = (const float*)d_in[10];
  const float* bng  = (const float*)d_in[11];
  const float* bnb  = (const float*)d_in[12];
  const float* ln1g = (const float*)d_in[13];
  const float* ln1b = (const float*)d_in[14];
  const float* ln2g = (const float*)d_in[15];
  const float* ln2b = (const float*)d_in[16];
  float* out = (float*)d_out;

  // workspace carve-up (~47 MB); attn/h/z alias q/k/v after those are consumed
  char* ws = (char*)d_ws;
  size_t o = 0;
  auto take = [&](size_t bytes) -> void* {
    void* p = ws + o;
    o += (bytes + 255) & ~(size_t)255;
    return p;
  };
  const size_t ND = (size_t)N_TOK * DMODEL;
  float* q    = (float*)take(ND * 4);
  float* k    = (float*)take(ND * 4);
  float* v    = (float*)take(ND * 4);
  bf16* qb    = (bf16*)take(ND * 2);
  bf16* kb    = (bf16*)take(ND * 2);
  bf16* vT    = (bf16*)take(ND * 2);
  bf16* xb    = (bf16*)take(ND * 2);
  bf16* lhb   = (bf16*)take(ND * 2);
  bf16* wqb   = (bf16*)take((size_t)DMODEL * DMODEL * 2);
  bf16* wkb   = (bf16*)take((size_t)DMODEL * DMODEL * 2);
  bf16* wvb   = (bf16*)take((size_t)DMODEL * DMODEL * 2);
  bf16* m1wb  = (bf16*)take((size_t)HDIM * DMODEL * 2);
  bf16* m2wb  = (bf16*)take((size_t)DMODEL * HDIM * 2);
  float* stats = (float*)take((2 + 2 * HDIM) * 4);
  float* sq = stats, *sk = stats + 1;
  float* bnscale = stats + 2, *bnshift = stats + 2 + HDIM;
  float* attn = q;  // q dead after qb+sumsq
  float* h    = k;  // k dead after kb+sumsq
  float* z    = v;  // v dead after transpose

  const int nd = (int)ND;
  dim3 b256(256);

  // 1) bf16 conversions
  k_f32_to_bf16<<<(nd + 255) / 256, b256, 0, stream>>>(x, xb, nd);
  k_f32_to_bf16<<<(DMODEL * DMODEL + 255) / 256, b256, 0, stream>>>(wq, wqb, DMODEL * DMODEL);
  k_f32_to_bf16<<<(DMODEL * DMODEL + 255) / 256, b256, 0, stream>>>(wk, wkb, DMODEL * DMODEL);
  k_f32_to_bf16<<<(DMODEL * DMODEL + 255) / 256, b256, 0, stream>>>(wv, wvb, DMODEL * DMODEL);
  k_f32_to_bf16<<<(HDIM * DMODEL + 255) / 256, b256, 0, stream>>>(m1w, m1wb, HDIM * DMODEL);
  k_f32_to_bf16<<<(DMODEL * HDIM + 255) / 256, b256, 0, stream>>>(m2w, m2wb, DMODEL * HDIM);

  // 2) projections
  k_proj<<<N_TOK / 16, b256, 0, stream>>>(xb, wqb, bq, q);
  k_proj<<<N_TOK / 16, b256, 0, stream>>>(xb, wkb, bk, k);
  k_proj<<<N_TOK / 16, b256, 0, stream>>>(xb, wvb, bv, v);

  // 3) Frobenius norms
  k_zero_f32<<<1, b256, 0, stream>>>(sq, 2);
  k_sumsq<<<512, b256, 0, stream>>>(q, nd, sq);
  k_sumsq<<<512, b256, 0, stream>>>(k, nd, sk);

  // 4) q,k -> bf16; v -> v^T bf16
  k_f32_to_bf16<<<(nd + 255) / 256, b256, 0, stream>>>(q, qb, nd);
  k_f32_to_bf16<<<(nd + 255) / 256, b256, 0, stream>>>(k, kb, nd);
  k_transpose_bf16<<<dim3(N_TOK / 32, DMODEL / 32), b256, 0, stream>>>(v, vT);

  // 5) fused conv-attention (the two 34-GFLOP GEMMs + 3x3 conv, A stays in LDS)
  k_fused_attn<<<N_TOK / 64, b256, 0, stream>>>(qb, kb, vT, sq, sk, cw, cb, attn);

  // 6) residual + double layernorm
  k_ln<<<N_TOK, b256, 0, stream>>>(x, attn, ln1g, ln1b, ln2g, ln2b, h, lhb);

  // 7) MLP1
  k_mlp1<<<N_TOK / 16, b256, 0, stream>>>(lhb, m1wb, z);

  // 8) batchnorm stats
  k_bnstats<<<HDIM, b256, 0, stream>>>(z, bng, bnb, bnscale, bnshift);

  // 9) bn + relu + MLP2 + residual -> out
  k_mlp2<<<N_TOK / 16, b256, 0, stream>>>(z, h, m2wb, bnscale, bnshift, out);
}
